// GCNNeck_24962349924890
// MI455X (gfx1250) — matrix operand order repaired and checked by reading, not compiled, not run on hardware
//
#include <hip/hip_runtime.h>

typedef __attribute__((ext_vector_type(16))) _Float16 v16h;
typedef __attribute__((ext_vector_type(8)))  _Float16 v8h;
typedef __attribute__((ext_vector_type(8)))  float    v8f;

#define N_PTS 4096
#define DIM   256
#define RAD2  (0.125f * 0.125f)

// workspace layout (bytes), all 16B-aligned
#define OFF_A1H    0u        // 256*256 f16  (W1a - W1b)       : 131072
#define OFF_W1BH   131072u   // 256*256 f16  W1b               : 131072
#define OFF_A2H    262144u   // 256*256 f16  (W2a - W2b)       : 131072
#define OFF_DEG    393216u   // 4096 f32 degrees               : 16384
#define OFF_XH     409600u   // 4096*256 f16 descriptors       : 2097152
#define OFF_SH     2506752u  // 4096*256 f16 S = M @ X         : 2097152
#define OFF_PREH   4603904u  // 4096*256 f16 pre               : 2097152
// total ~6.4 MB

// ---------- weight prep: A1 = W1[:,:D]-W1[:,D:], W1b, A2 = W2[:,:D]-W2[:,D:] ----------
__global__ void prep_weights(const float* __restrict__ W1, const float* __restrict__ W2,
                             _Float16* __restrict__ A1h, _Float16* __restrict__ W1bh,
                             _Float16* __restrict__ A2h) {
    int idx = blockIdx.x * blockDim.x + threadIdx.x;   // 0..65535
    int r = idx >> 8, k = idx & 255;
    float w1a = W1[r * 512 + k];
    float w1b = W1[r * 512 + 256 + k];
    A1h[idx]  = (_Float16)(w1a - w1b);
    W1bh[idx] = (_Float16)w1b;
    float w2a = W2[r * 512 + k];
    float w2b = W2[r * 512 + 256 + k];
    A2h[idx]  = (_Float16)(w2a - w2b);                 // A2h[c*256+r] = W2[c,r]-W2[c,256+r]
}

// ---------- descriptor prep: copy fp32 into out[:, :256], f16 copy for WMMA ----------
__global__ void prep_desc(const float* __restrict__ X, float* __restrict__ out,
                          _Float16* __restrict__ Xh) {
    int idx = blockIdx.x * blockDim.x + threadIdx.x;   // 0 .. 4096*256-1
    int i = idx >> 8, k = idx & 255;
    float d = X[idx];
    out[i * 512 + k] = d;
    Xh[idx] = (_Float16)d;
}

// ---------- neighbor aggregation: deg[i], S[i,:] = sum_{j in N(i)} X[j,:] ----------
__global__ __launch_bounds__(256) void neighbor_agg(const float* __restrict__ centers,
                                                    const _Float16* __restrict__ Xh,
                                                    _Float16* __restrict__ Sh,
                                                    float* __restrict__ deg) {
    __shared__ float scx[N_PTS];
    __shared__ float scy[N_PTS];
    __shared__ float scz[N_PTS];
    int tid = threadIdx.x;
    for (int j = tid; j < N_PTS; j += 256) {
        scx[j] = centers[j * 3 + 0];
        scy[j] = centers[j * 3 + 1];
        scz[j] = centers[j * 3 + 2];
    }
    __syncthreads();

    int lane = tid & 31;
    int wave = tid >> 5;
    int i = blockIdx.x * 8 + wave;                     // one node per wave
    float cx = scx[i], cy = scy[i], cz = scz[i];

    float acc[8];
#pragma unroll
    for (int c = 0; c < 8; ++c) acc[c] = 0.0f;
    int degc = 0;

    for (int jb = 0; jb < N_PTS; jb += 32) {
        int j = jb + lane;
        float dx = scx[j] - cx;
        float dy = scy[j] - cy;
        float dz = scz[j] - cz;
        bool nb = (dx * dx + dy * dy + dz * dz <= RAD2) && (j != i);
        unsigned mask = __builtin_amdgcn_ballot_w32(nb);
        degc += __builtin_popcount(mask);
        while (mask) {
            int j0 = jb + __builtin_ctz(mask);
            mask &= mask - 1;
            v8h x = *(const v8h*)(Xh + j0 * DIM + lane * 8);  // lane owns 8 contiguous cols
#pragma unroll
            for (int c = 0; c < 8; ++c) acc[c] += (float)x[c];
        }
    }

    v8h s;
#pragma unroll
    for (int c = 0; c < 8; ++c) s[c] = (_Float16)acc[c];
    *(v8h*)(Sh + i * DIM + lane * 8) = s;
    if (lane == 0) deg[i] = (float)degc;
}

// ---------- WMMA fragment loaders (row-major f16, ld = 256) ----------
__device__ inline v16h load_fragA(const _Float16* __restrict__ M, int row0, int k0, int lane) {
    // 16x32 A tile: lane holds row (lane&15); k runs [kb,kb+8) and [kb+16,kb+24)
    int m  = row0 + (lane & 15);
    int kb = k0 + ((lane & 16) ? 8 : 0);
    const _Float16* p = M + m * DIM + kb;
    v8h lo = *(const v8h*)p;
    v8h hi = *(const v8h*)(p + 16);
    v16h a;
#pragma unroll
    for (int h = 0; h < 8; ++h) { a[h] = lo[h]; a[h + 8] = hi[h]; }
    return a;
}

__device__ inline v16h load_fragB(const _Float16* __restrict__ W, int n0, int k0, int lane) {
    // 32x16 B tile with B[k,n] = W[n*DIM + k]: lane holds col (lane&15);
    // k runs [kb,kb+8) and [kb+16,kb+24), kb = k0 + (lane&16 ? 8 : 0)
    int n  = n0 + (lane & 15);
    int kb = k0 + ((lane & 16) ? 8 : 0);
    const _Float16* p = W + n * DIM + kb;
    v8h lo = *(const v8h*)p;
    v8h hi = *(const v8h*)(p + 16);
    v16h b;
#pragma unroll
    for (int h = 0; h < 8; ++h) { b[h] = lo[h]; b[h + 8] = hi[h]; }
    return b;
}

// ---------- GEMM1: pre = S@A1^T + deg*(X@W1b^T + b1), stored f16 ----------
__global__ __launch_bounds__(256) void gemm1(const _Float16* __restrict__ Sh,
                                             const _Float16* __restrict__ Xh,
                                             const _Float16* __restrict__ A1h,
                                             const _Float16* __restrict__ W1bh,
                                             const float* __restrict__ b1,
                                             const float* __restrict__ deg,
                                             _Float16* __restrict__ preh) {
    int tid = threadIdx.x, lane = tid & 31, wave = tid >> 5;
    int tile = blockIdx.x * 8 + wave;                  // 4096 tiles: 256 (M) x 16 (N)
    int row0 = (tile >> 4) * 16;
    int n0   = (tile & 15) * 16;

    v8f acc1 = {};
    v8f acc2 = {};
#pragma unroll
    for (int kt = 0; kt < DIM; kt += 32) {
        v16h aS = load_fragA(Sh, row0, kt, lane);
        v16h bA = load_fragB(A1h, n0, kt, lane);
        acc1 = __builtin_amdgcn_wmma_f32_16x16x32_f16(false, aS, false, bA,
                                                      (short)0, acc1, false, false);
        v16h aX = load_fragA(Xh, row0, kt, lane);
        v16h bW = load_fragB(W1bh, n0, kt, lane);
        acc2 = __builtin_amdgcn_wmma_f32_16x16x32_f16(false, aX, false, bW,
                                                      (short)0, acc2, false, false);
    }

    int ncol = n0 + (lane & 15);
    int mbase = row0 + ((lane & 16) ? 8 : 0);
    float b1v = b1[ncol];
#pragma unroll
    for (int p = 0; p < 8; ++p) {
        int m = mbase + p;
        float pre = acc1[p] + deg[m] * (acc2[p] + b1v);
        preh[m * DIM + ncol] = (_Float16)pre;
    }
}

// ---------- GEMM2: pooled = (pre@A2^T + deg*b2)/N -> out[:, 256:512] ----------
__global__ __launch_bounds__(256) void gemm2(const _Float16* __restrict__ preh,
                                             const _Float16* __restrict__ A2h,
                                             const float* __restrict__ b2,
                                             const float* __restrict__ deg,
                                             float* __restrict__ out) {
    int tid = threadIdx.x, lane = tid & 31, wave = tid >> 5;
    int tile = blockIdx.x * 8 + wave;
    int row0 = (tile >> 4) * 16;
    int n0   = (tile & 15) * 16;

    v8f acc = {};
#pragma unroll
    for (int kt = 0; kt < DIM; kt += 32) {
        v16h a = load_fragA(preh, row0, kt, lane);
        v16h b = load_fragB(A2h, n0, kt, lane);
        acc = __builtin_amdgcn_wmma_f32_16x16x32_f16(false, a, false, b,
                                                     (short)0, acc, false, false);
    }

    int ncol = n0 + (lane & 15);
    int mbase = row0 + ((lane & 16) ? 8 : 0);
    float b2v = b2[ncol];
    const float invn = 1.0f / (float)N_PTS;
#pragma unroll
    for (int p = 0; p < 8; ++p) {
        int m = mbase + p;
        out[m * 512 + 256 + ncol] = (acc[p] + deg[m] * b2v) * invn;
    }
}

extern "C" void kernel_launch(void* const* d_in, const int* in_sizes, int n_in,
                              void* d_out, int out_size, void* d_ws, size_t ws_size,
                              hipStream_t stream) {
    const float* centers = (const float*)d_in[0];   // 4096*3
    const float* X       = (const float*)d_in[1];   // 4096*256
    const float* W1      = (const float*)d_in[2];   // 256*512
    const float* b1      = (const float*)d_in[3];   // 256
    const float* W2      = (const float*)d_in[4];   // 256*512
    const float* b2      = (const float*)d_in[5];   // 256
    float* out = (float*)d_out;                     // 4096*512

    char* ws = (char*)d_ws;
    _Float16* A1h  = (_Float16*)(ws + OFF_A1H);
    _Float16* W1bh = (_Float16*)(ws + OFF_W1BH);
    _Float16* A2h  = (_Float16*)(ws + OFF_A2H);
    float*    deg  = (float*)   (ws + OFF_DEG);
    _Float16* Xh   = (_Float16*)(ws + OFF_XH);
    _Float16* Sh   = (_Float16*)(ws + OFF_SH);
    _Float16* preh = (_Float16*)(ws + OFF_PREH);

    prep_weights<<<256, 256, 0, stream>>>(W1, W2, A1h, W1bh, A2h);
    prep_desc<<<(N_PTS * DIM) / 256, 256, 0, stream>>>(X, out, Xh);
    neighbor_agg<<<N_PTS / 8, 256, 0, stream>>>(centers, Xh, Sh, deg);
    gemm1<<<512, 256, 0, stream>>>(Sh, Xh, A1h, W1bh, b1, deg, preh);
    gemm2<<<512, 256, 0, stream>>>(preh, A2h, b2, deg, out);
}
